// SubLevelGraphFiltration_69200513073319
// MI455X (gfx1250) — matrix operand order repaired and checked by reading, not compiled, not run on hardware
//
#include <hip/hip_runtime.h>
#include <hip/hip_bf16.h>
#include <math.h>

#define NN 50000
#define NE 600000
#define ZLD 640
#define BN_CHUNK 512
#define NCHUNK 98   // ceil(50000/512)
#define NBLKY 128   // GEMM grid.y (blocks of 4 waves per n-tile column)

typedef _Float16 h16;
typedef __attribute__((ext_vector_type(16))) _Float16 v16h;
typedef __attribute__((ext_vector_type(8)))  _Float16 h8;
typedef __attribute__((ext_vector_type(4)))  _Float16 h4;
typedef __attribute__((ext_vector_type(8)))  float    v8f;

// ---------------- small utility kernels ----------------

__global__ void k_f32_to_f16(const float* __restrict__ s, h16* __restrict__ d, int n) {
    int i = blockIdx.x * blockDim.x + threadIdx.x;
    if (i < n) d[i] = (h16)s[i];
}

__global__ void k_zero_f32(float* __restrict__ p, int n) {
    int i = blockIdx.x * blockDim.x + threadIdx.x;
    if (i < n) p[i] = 0.0f;
}

// h[n, 0:128] = embed_deg[deg[n]], h[n,128:256] = embed_lab[lab[n]]  (f16 into Zh slice)
__global__ void k_embed(const int* __restrict__ deg, const int* __restrict__ lab,
                        const float* __restrict__ ed, const float* __restrict__ el,
                        h16* __restrict__ Zh) {
    int i = blockIdx.x * blockDim.x + threadIdx.x;
    if (i >= NN * 256) return;
    int n = i >> 8, f = i & 255;
    float v = (f < 128) ? ed[deg[n] * 128 + f] : el[lab[n] * 128 + (f - 128)];
    Zh[(size_t)n * ZLD + f] = (h16)v;
}

// agg[dst] += x[src], edge-parallel, 4 features per thread
__global__ void k_scatter(const int* __restrict__ ei, const h16* __restrict__ Zh, int zoff,
                          float* __restrict__ agg, int din, int shift) {
    long i = (long)blockIdx.x * blockDim.x + threadIdx.x;
    if (i >= ((long)NE << shift)) return;
    int e = (int)(i >> shift);
    int g = ((int)i & ((1 << shift) - 1)) << 2;
    int s = ei[e];
    int d = ei[NE + e];
    h4 x = *(const h4*)(Zh + (size_t)s * ZLD + zoff + g);
    float* ap = agg + (size_t)d * din + g;
    atomicAdd(ap + 0, (float)x[0]);
    atomicAdd(ap + 1, (float)x[1]);
    atomicAdd(ap + 2, (float)x[2]);
    atomicAdd(ap + 3, (float)x[3]);
}

// yh = f16( (1+eps[l]) * x + agg )
__global__ void k_combine(const h16* __restrict__ Zh, int zoff, const float* __restrict__ agg,
                          const float* __restrict__ eps, int l, h16* __restrict__ yh, int dshift) {
    int i = blockIdx.x * blockDim.x + threadIdx.x;
    if (i >= (NN << dshift)) return;
    int n = i >> dshift, f = i & ((1 << dshift) - 1);
    float x = (float)Zh[(size_t)n * ZLD + zoff + f];
    float v = (1.0f + eps[l]) * x + agg[i];
    yh[i] = (h16)v;
}

// ---------------- WMMA GEMM:  C[M,N] = A[M,K](f16) * B[K,N](f16) + bias ----------------
// Persistent waves: each wave owns one 16-wide n-tile, preloads ALL B fragments for
// that column strip into VGPRs (KSTEPS x 8 VGPRs), then streams m-tiles:
// steady state = two 16B A loads + one v_wmma_f32_16x16x32_f16 per K-slice.
// MODE 0: store raw f32.  MODE 1: LeakyReLU then store f16.
template <int KSTEPS, int MODE>
__global__ void k_gemm_wmma(const h16* __restrict__ A, int lda,
                            const h16* __restrict__ B, int N,
                            const float* __restrict__ bias,
                            float* __restrict__ Cf, h16* __restrict__ Ch, int ldc, int M) {
    int wave = threadIdx.x >> 5;
    int lane = threadIdx.x & 31;
    int nt = blockIdx.x;
    int l16 = lane & 15;
    int hi  = lane >> 4;
    int col = nt * 16 + l16;

    // B fragment (contiguous K, same order as A): col = lane&15 (both halves), K = 16*(lane>>4) + e within a 32-K slice
    v16h bfrag[KSTEPS];
    {
        const h16* Bp = B + (size_t)(hi * 16) * N + col;
#pragma unroll
        for (int ks = 0; ks < KSTEPS; ks++)
#pragma unroll
            for (int e = 0; e < 16; e++)
                bfrag[ks][e] = Bp[(size_t)(ks * 32 + e) * N];
    }
    float bia = bias[col];

    int mtiles = M >> 4;
    int stride = gridDim.y * (blockDim.x >> 5);
    // loop bound is wave-uniform: EXEC stays all-ones for WMMA
    for (int mt = blockIdx.y * (blockDim.x >> 5) + wave; mt < mtiles; mt += stride) {
        // A-layout: row = lane&15, K = 16*(lane>>4) + e within a 32-K slice (matches B)
        const h16* Arow = A + (size_t)(mt * 16 + l16) * lda + hi * 16;
        // speculative prefetch of this wave's next m-tile (OOB prefetch is silently dropped)
        __builtin_prefetch(Arow + (size_t)stride * 16 * lda, 0, 1);

        v8f c = {};
#pragma unroll
        for (int ks = 0; ks < KSTEPS; ks++) {
            h8 alo = *(const h8*)(Arow + ks * 32);
            h8 ahi = *(const h8*)(Arow + ks * 32 + 8);
            v16h a;
#pragma unroll
            for (int e = 0; e < 8; e++) { a[e] = alo[e]; a[e + 8] = ahi[e]; }
            c = __builtin_amdgcn_wmma_f32_16x16x32_f16(false, a, false, bfrag[ks],
                                                       (short)0, c, false, false);
        }

        // D-layout: lane (hi,l16), vgpr v -> (M = v + 8*hi, N = l16)
#pragma unroll
        for (int v = 0; v < 8; v++) {
            int row = mt * 16 + v + hi * 8;
            float val = c[v] + bia;
            if (MODE == 1) {
                val = (val > 0.0f) ? val : 0.01f * val;
                Ch[(size_t)row * ldc + col] = (h16)val;
            } else {
                Cf[(size_t)row * ldc + col] = val;
            }
        }
    }
}

// ---------------- BatchNorm (training mode, biased variance) ----------------

__global__ void k_bn_partial(const float* __restrict__ u, float* __restrict__ ps,
                             float* __restrict__ ps2, int M) {
    int c = threadIdx.x;          // 128 columns
    int chunk = blockIdx.x;
    int r0 = chunk * BN_CHUNK;
    int r1 = r0 + BN_CHUNK; if (r1 > M) r1 = M;
    float s = 0.0f, s2 = 0.0f;
    for (int r = r0; r < r1; r++) {
        float x = u[(size_t)r * 128 + c];
        s += x; s2 += x * x;
    }
    ps[chunk * 128 + c] = s;
    ps2[chunk * 128 + c] = s2;
}

__global__ void k_bn_finalize(const float* __restrict__ ps, const float* __restrict__ ps2,
                              int nchunk, const float* __restrict__ g, const float* __restrict__ b,
                              float* __restrict__ bnA, float* __restrict__ bnB, int M) {
    int c = threadIdx.x;
    float s = 0.0f, s2 = 0.0f;
    for (int k = 0; k < nchunk; k++) { s += ps[k * 128 + c]; s2 += ps2[k * 128 + c]; }
    float mean = s / (float)M;
    float var  = s2 / (float)M - mean * mean;
    float rstd = rsqrtf(var + 1e-5f);
    float Aa = g[c] * rstd;
    bnA[c] = Aa;
    bnB[c] = b[c] - mean * Aa;
}

// out = f16(lrelu(u * bnA + bnB)) into f16 slice [ld=ldo, col offset ooff]
__global__ void k_bn_apply(const float* __restrict__ u, const float* __restrict__ bnA,
                           const float* __restrict__ bnB, h16* __restrict__ outH,
                           int ldo, int ooff) {
    int i = blockIdx.x * blockDim.x + threadIdx.x;
    if (i >= NN * 128) return;
    int n = i >> 7, f = i & 127;
    float v = u[i] * bnA[f] + bnB[f];
    v = (v > 0.0f) ? v : 0.01f * v;
    outH[(size_t)n * ldo + ooff + f] = (h16)v;
}

// ---------------- final 128 -> 1 + sigmoid, wave per node ----------------
__global__ void k_final(const h16* __restrict__ xh, const float* __restrict__ w2,
                        const float* __restrict__ b2, float* __restrict__ out) {
    int wid = (int)((blockIdx.x * (long)blockDim.x + threadIdx.x) >> 5);
    int lane = threadIdx.x & 31;
    if (wid >= NN) return;
    h4 x = *(const h4*)(xh + (size_t)wid * 128 + lane * 4);
    float4 w = *(const float4*)(w2 + lane * 4);
    float s = (float)x[0] * w.x + (float)x[1] * w.y + (float)x[2] * w.z + (float)x[3] * w.w;
#pragma unroll
    for (int o = 16; o > 0; o >>= 1) s += __shfl_xor(s, o, 32);
    if (lane == 0) out[wid] = 1.0f / (1.0f + expf(-(s + b2[0])));
}

// ---------------- driver ----------------
extern "C" void kernel_launch(void* const* d_in, const int* in_sizes, int n_in,
                              void* d_out, int out_size, void* d_ws, size_t ws_size,
                              hipStream_t stream) {
    const int*   node_deg   = (const int*)d_in[0];
    const int*   node_lab   = (const int*)d_in[1];
    const int*   edge_index = (const int*)d_in[2];
    const float* embed_deg  = (const float*)d_in[3];
    const float* embed_lab  = (const float*)d_in[4];
    const float* eps        = (const float*)d_in[5];
    const float* conv_w1[3] = {(const float*)d_in[6],  (const float*)d_in[7],  (const float*)d_in[8]};
    const float* conv_b1[3] = {(const float*)d_in[9],  (const float*)d_in[10], (const float*)d_in[11]};
    const float* conv_w2[3] = {(const float*)d_in[12], (const float*)d_in[13], (const float*)d_in[14]};
    const float* conv_b2[3] = {(const float*)d_in[15], (const float*)d_in[16], (const float*)d_in[17]};
    const float* bn_g[3]    = {(const float*)d_in[18], (const float*)d_in[19], (const float*)d_in[20]};
    const float* bn_b[3]    = {(const float*)d_in[21], (const float*)d_in[22], (const float*)d_in[23]};
    const float* fc_w1   = (const float*)d_in[24];
    const float* fc_b1   = (const float*)d_in[25];
    const float* fc_bn_g = (const float*)d_in[26];
    const float* fc_bn_b = (const float*)d_in[27];
    const float* fc_w2   = (const float*)d_in[28];
    const float* fc_b2   = (const float*)d_in[29];

    // workspace layout (bytes)
    char* ws = (char*)d_ws;
    h16*   Zh  = (h16*)ws;                      // [NN,640] f16  : 64,000,000
    float* SA  = (float*)(ws + 64000000);       // agg / gemm2-out f32, up to [NN,256] : 51,200,000
    h16*   yh  = (h16*)(ws + 115200000);        // [NN,256] f16 : 25,600,000
    h16*   th  = (h16*)(ws + 140800000);        // [NN,256] f16 : 25,600,000
    h16*   wh  = (h16*)(ws + 166400000);        // converted weights, ~492 KB
    float* ps  = (float*)(ws + 167000000);      // [NCHUNK,128]
    float* ps2 = (float*)(ws + 167100000);      // [NCHUNK,128]
    float* bnA = (float*)(ws + 167200000);      // [128]
    float* bnB = (float*)(ws + 167201024);      // [128]

    h16* w1h[3] = { wh,          wh + 65536,  wh + 81920  };
    h16* w2h[3] = { wh + 98304,  wh + 131072, wh + 147456 };
    h16* fcw1h  = wh + 163840;

    auto cvt = [&](const float* s, h16* d, int n) {
        k_f32_to_f16<<<(n + 255) / 256, 256, 0, stream>>>(s, d, n);
    };
    cvt(conv_w1[0], w1h[0], 256 * 256);
    cvt(conv_w1[1], w1h[1], 128 * 128);
    cvt(conv_w1[2], w1h[2], 128 * 128);
    cvt(conv_w2[0], w2h[0], 256 * 128);
    cvt(conv_w2[1], w2h[1], 128 * 128);
    cvt(conv_w2[2], w2h[2], 128 * 128);
    cvt(fc_w1, fcw1h, 640 * 128);

    // node embedding -> Zh[:, 0:256]
    k_embed<<<(NN * 256 + 255) / 256, 256, 0, stream>>>(node_deg, node_lab, embed_deg, embed_lab, Zh);

    const int DIN[3]  = {256, 128, 128};
    const int ZOFF[4] = {0, 256, 384, 512};

    for (int l = 0; l < 3; l++) {
        int din  = DIN[l];
        int zin  = ZOFF[l];
        int zout = ZOFF[l + 1];
        int na = NN * din;

        k_zero_f32<<<(na + 255) / 256, 256, 0, stream>>>(SA, na);

        int shift = (din == 256) ? 6 : 5;   // din/4 groups per edge
        long nthr = (long)NE << shift;
        k_scatter<<<(int)((nthr + 255) / 256), 256, 0, stream>>>(edge_index, Zh, zin, SA, din, shift);

        int dshift = (din == 256) ? 8 : 7;
        k_combine<<<(na + 255) / 256, 256, 0, stream>>>(Zh, zin, SA, eps, l, yh, dshift);

        // GEMM1: lrelu(yh @ W1 + b1) -> th (f16)
        dim3 g1(din / 16, NBLKY);
        if (din == 256)
            k_gemm_wmma<8, 1><<<g1, 128, 0, stream>>>(yh, din, w1h[l], din, conv_b1[l],
                                                      nullptr, th, din, NN);
        else
            k_gemm_wmma<4, 1><<<g1, 128, 0, stream>>>(yh, din, w1h[l], din, conv_b1[l],
                                                      nullptr, th, din, NN);
        // GEMM2: th @ W2 + b2 -> SA (f32)
        dim3 g2(128 / 16, NBLKY);
        if (din == 256)
            k_gemm_wmma<8, 0><<<g2, 128, 0, stream>>>(th, din, w2h[l], 128, conv_b2[l],
                                                      SA, nullptr, 128, NN);
        else
            k_gemm_wmma<4, 0><<<g2, 128, 0, stream>>>(th, din, w2h[l], 128, conv_b2[l],
                                                      SA, nullptr, 128, NN);
        // BN + lrelu -> Zh[:, zout:zout+128]
        k_bn_partial<<<NCHUNK, 128, 0, stream>>>(SA, ps, ps2, NN);
        k_bn_finalize<<<1, 128, 0, stream>>>(ps, ps2, NCHUNK, bn_g[l], bn_b[l], bnA, bnB, NN);
        k_bn_apply<<<(NN * 128 + 255) / 256, 256, 0, stream>>>(SA, bnA, bnB, Zh, ZLD, zout);
    }

    // FC: Zh[N,640] @ fc_w1 + b1 -> SA (f32), BN + lrelu -> yh (f16, ld=128)
    dim3 gf(128 / 16, NBLKY);
    k_gemm_wmma<20, 0><<<gf, 128, 0, stream>>>(Zh, ZLD, fcw1h, 128, fc_b1, SA, nullptr, 128, NN);
    k_bn_partial<<<NCHUNK, 128, 0, stream>>>(SA, ps, ps2, NN);
    k_bn_finalize<<<1, 128, 0, stream>>>(ps, ps2, NCHUNK, fc_bn_g, fc_bn_b, bnA, bnB, NN);
    k_bn_apply<<<(NN * 128 + 255) / 256, 256, 0, stream>>>(SA, bnA, bnB, yh, 128, 0);

    // sigmoid(x @ fc_w2 + b2) -> out[N]
    long fthreads = (long)NN * 32;
    k_final<<<(int)((fthreads + 255) / 256), 256, 0, stream>>>(yh, fc_w2, fc_b2, (float*)d_out);
}
